// MoELayer_71193377898542
// MI455X (gfx1250) — matrix operand order repaired and checked by reading, not compiled, not run on hardware
//
#include <hip/hip_runtime.h>
#include <stdint.h>

// Problem constants (match reference)
#define NTOK 8192
#define DIN  1024
#define HID  1024
#define NEXP 8
#define TOPK 2

typedef __attribute__((ext_vector_type(16))) __bf16 v16bf;
typedef __attribute__((ext_vector_type(8)))  __bf16 v8bf;
typedef __attribute__((ext_vector_type(8)))  float  v8f;

#define KP 40   // padded LDS row length (bf16 elems) for a 32-wide K tile

// ---------------------------------------------------------------------------
// CDNA5 async global->LDS copy (ASYNCcnt-tracked, no VGPR round trip).
// vdst = 32-bit LDS byte address, vaddr = 64-bit global address (GV mode).
template <int OFF>
__device__ __forceinline__ void async_ld_b128(uint32_t lds, const void* g) {
  if constexpr (OFF == 0) {
    asm volatile("global_load_async_to_lds_b128 %0, %1, off"
                 :: "v"(lds), "v"(g) : "memory");
  } else {
    asm volatile("global_load_async_to_lds_b128 %0, %1, off offset:%2"
                 :: "v"(lds), "v"(g), "n"(OFF) : "memory");
  }
}
#define WAIT_ASYNC(N) asm volatile("s_wait_asynccnt " #N ::: "memory")

__device__ __forceinline__ uint32_t lds_addr(const void* p) {
  // Generic pointers to LDS carry the wave-relative LDS offset in the low 32b.
  return (uint32_t)(uintptr_t)p;
}

// ---------------------------------------------------------------------------
// Zero output + expert counters
__global__ void moe_zero(float* __restrict__ out, int* __restrict__ counts) {
  int i = blockIdx.x * 256 + threadIdx.x;
  if (i < NTOK * HID) out[i] = 0.0f;
  if (i < NEXP) counts[i] = 0;
}

// ---------------------------------------------------------------------------
// Convert activations x [N,D] f32 -> bf16 (4 elems/thread)
__global__ void conv_x(const float* __restrict__ x, __bf16* __restrict__ xb) {
  int base = (blockIdx.x * 256 + threadIdx.x) * 4;
  float4 v = *(const float4*)(x + base);
  xb[base + 0] = (__bf16)v.x;
  xb[base + 1] = (__bf16)v.y;
  xb[base + 2] = (__bf16)v.z;
  xb[base + 3] = (__bf16)v.w;
}

// ---------------------------------------------------------------------------
// Transpose-convert weights: src [E][R][C] f32 -> dst [E][C][R] bf16
__global__ void conv_w(const float* __restrict__ src, __bf16* __restrict__ dst,
                       int R, int C) {
  __shared__ float tile[32][33];
  int e = blockIdx.z;
  int c0 = blockIdx.x * 32, r0 = blockIdx.y * 32;
  int tx = threadIdx.x & 31, ty = threadIdx.x >> 5;   // 32x8
  const float* s = src + (size_t)e * R * C;
  __bf16*      d = dst + (size_t)e * R * C;
  #pragma unroll
  for (int i = 0; i < 32; i += 8)
    tile[ty + i][tx] = s[(size_t)(r0 + ty + i) * C + c0 + tx];
  __syncthreads();
  #pragma unroll
  for (int i = 0; i < 32; i += 8)
    d[(size_t)(c0 + ty + i) * R + r0 + tx] = (__bf16)tile[tx][ty + i];
}

// ---------------------------------------------------------------------------
// Gating: wave-per-token logits, top-2, softmax, append to expert lists
__global__ __launch_bounds__(256) void moe_gate(
    const float* __restrict__ x, const float* __restrict__ Wg,
    const float* __restrict__ bg, int* __restrict__ counts,
    int* __restrict__ tokens, float* __restrict__ gates) {
  int wave = threadIdx.x >> 5;
  int lane = threadIdx.x & 31;
  int n = blockIdx.x * 8 + wave;

  float acc[NEXP];
  #pragma unroll
  for (int e = 0; e < NEXP; e++) acc[e] = 0.0f;

  const float* xr = x + (size_t)n * DIN;
  for (int d = lane; d < DIN; d += 32) {
    float xv = xr[d];
    #pragma unroll
    for (int e = 0; e < NEXP; e++) acc[e] += xv * Wg[e * DIN + d];
  }
  #pragma unroll
  for (int o = 16; o > 0; o >>= 1) {
    #pragma unroll
    for (int e = 0; e < NEXP; e++) acc[e] += __shfl_down(acc[e], o, 32);
  }
  if (lane == 0) {
    float lg[NEXP];
    #pragma unroll
    for (int e = 0; e < NEXP; e++) lg[e] = acc[e] + bg[e];
    // top-2, first-occurrence wins ties (matches lax.top_k)
    int i0 = 0;
    #pragma unroll
    for (int e = 1; e < NEXP; e++) if (lg[e] > lg[i0]) i0 = e;
    int i1 = -1;
    #pragma unroll
    for (int e = 0; e < NEXP; e++)
      if (e != i0 && (i1 < 0 || lg[e] > lg[i1])) i1 = e;
    float m = fmaxf(lg[i0], lg[i1]);
    float e0 = __expf(lg[i0] - m), e1 = __expf(lg[i1] - m);
    float inv = 1.0f / (e0 + e1);
    int p0 = atomicAdd(&counts[i0], 1);
    tokens[i0 * NTOK + p0] = n;  gates[i0 * NTOK + p0] = e0 * inv;
    int p1 = atomicAdd(&counts[i1], 1);
    tokens[i1 * NTOK + p1] = n;  gates[i1 * NTOK + p1] = e1 * inv;
  }
}

// ---------------------------------------------------------------------------
__device__ __forceinline__ v16bf cat16(v8bf lo, v8bf hi) {
  return __builtin_shufflevector(lo, hi, 0,1,2,3,4,5,6,7,8,9,10,11,12,13,14,15);
}

// ---------------------------------------------------------------------------
// GEMM1: h1[e][slot][h] = relu( xb[tok] @ W1t[e]^T + b1[e] )   (bf16 out)
// Double-buffered LDS tiles filled by async global->LDS copies; the pipeline
// is unrolled 2x so buffer indices are compile-time constants.
__global__ __launch_bounds__(256) void moe_gemm1(
    const __bf16* __restrict__ xb, const __bf16* __restrict__ w1t,
    const float* __restrict__ b1, const int* __restrict__ counts,
    const int* __restrict__ tokens, __bf16* __restrict__ h1) {
  int e  = blockIdx.y;
  int cnt = counts[e];
  int m0 = blockIdx.z * 128;
  if (m0 >= cnt) return;
  int n0 = blockIdx.x * 128;

  __shared__ __align__(16) __bf16 As[2][128][KP];
  __shared__ __align__(16) __bf16 Bs[2][128][KP];
  __shared__ int tokLds[128];

  int tid = threadIdx.x;
  int lane = tid & 31, wave = tid >> 5;
  int waveM = wave & 3, waveN = wave >> 2;
  int l15 = lane & 15, half = lane >> 4;

  if (tid < 128) {
    // clamp to a valid slot: clamped rows only feed discarded outputs,
    // keeping the async-load path branch-free
    int slot = m0 + tid;
    tokLds[tid] = tokens[e * NTOK + (slot < cnt ? slot : cnt - 1)];
  }
  __syncthreads();

  v8f acc[2][4];
  #pragma unroll
  for (int i = 0; i < 2; i++)
    #pragma unroll
    for (int j = 0; j < 4; j++) acc[i][j] = (v8f)(0.0f);

  const __bf16* Bbase = w1t + (size_t)e * HID * DIN + (size_t)n0 * DIN;

  // loader assignment: 512 16B chunks per tile, 2 per thread per tile
  int row0 = tid >> 2, row1 = row0 + 64;
  int kc = (tid & 3) * 8;
  const __bf16* gA0 = xb + (size_t)tokLds[row0] * DIN + kc;
  const __bf16* gA1 = xb + (size_t)tokLds[row1] * DIN + kc;
  const __bf16* gB0 = Bbase + (size_t)row0 * DIN + kc;
  const __bf16* gB1 = Bbase + (size_t)row1 * DIN + kc;
  uint32_t lA0[2], lA1[2], lB0[2], lB1[2];
  #pragma unroll
  for (int b = 0; b < 2; b++) {
    lA0[b] = lds_addr(&As[b][row0][kc]);
    lA1[b] = lds_addr(&As[b][row1][kc]);
    lB0[b] = lds_addr(&Bs[b][row0][kc]);
    lB1[b] = lds_addr(&Bs[b][row1][kc]);
  }

  auto issue = [&](auto bufc, auto offc) {  // bufc/offc: integral_constant
    constexpr int B = decltype(bufc)::value;
    constexpr int O = decltype(offc)::value;    // byte offset
    async_ld_b128<O>(lA0[B], gA0);  async_ld_b128<O>(lA1[B], gA1);
    async_ld_b128<O>(lB0[B], gB0);  async_ld_b128<O>(lB1[B], gB1);
  };
  auto advance = [&]() { gA0 += 64; gA1 += 64; gB0 += 64; gB1 += 64; };

  auto compute = [&](auto bufc) {
    constexpr int B = decltype(bufc)::value;
    v16bf afrag[2], bfrag[4];
    #pragma unroll
    for (int mf = 0; mf < 2; mf++) {
      int row = waveM * 32 + mf * 16 + l15;
      int kb = half * 8;   // A layout: lanes 0-15 K[0..7]+[16..23]; 16-31 K[8..15]+[24..31]
      afrag[mf] = cat16(*(const v8bf*)(&As[B][row][kb]),
                        *(const v8bf*)(&As[B][row][kb + 16]));
    }
    #pragma unroll
    for (int nf = 0; nf < 4; nf++) {
      int col = waveN * 64 + nf * 16 + l15;
      int kb = half * 8;   // B layout: lanes 0-15 K[0..7]+[16..23]; 16-31 K[8..15]+[24..31]
      bfrag[nf] = cat16(*(const v8bf*)(&Bs[B][col][kb]),
                        *(const v8bf*)(&Bs[B][col][kb + 16]));
    }
    #pragma unroll
    for (int mf = 0; mf < 2; mf++)
      #pragma unroll
      for (int nf = 0; nf < 4; nf++)
        acc[mf][nf] = __builtin_amdgcn_wmma_f32_16x16x32_bf16(
            false, afrag[mf], false, bfrag[nf], (short)0, acc[mf][nf],
            false, false);
  };

  using c0 = std::integral_constant<int, 0>;
  using c1 = std::integral_constant<int, 1>;
  using o0 = std::integral_constant<int, 0>;
  using o64 = std::integral_constant<int, 64>;

  // prologue: stages 0 (buf0) and 1 (buf1)
  issue(c0{}, o0{});
  issue(c1{}, o64{});
  advance();

  // steady state: (DIN/32 - 2) / 2 double-stages
  for (int it = 0; it < (DIN / 32 - 2) / 2; ++it) {
    WAIT_ASYNC(4);  __syncthreads();
    compute(c0{});
    __syncthreads();
    issue(c0{}, o0{});
    WAIT_ASYNC(4);  __syncthreads();
    compute(c1{});
    __syncthreads();
    issue(c1{}, o64{});
    advance();
  }
  // drain last two stages
  WAIT_ASYNC(4);  __syncthreads();
  compute(c0{});
  __syncthreads();
  WAIT_ASYNC(0);  __syncthreads();
  compute(c1{});

  // epilogue: bias + relu -> bf16 h1
  #pragma unroll
  for (int mf = 0; mf < 2; mf++) {
    int sbase = m0 + waveM * 32 + mf * 16 + half * 8;
    #pragma unroll
    for (int nf = 0; nf < 4; nf++) {
      int h = n0 + waveN * 64 + nf * 16 + l15;
      float bv = b1[e * HID + h];
      #pragma unroll
      for (int r = 0; r < 8; r++) {
        int s = sbase + r;
        if (s < cnt) {
          float v = fmaxf(acc[mf][nf][r] + bv, 0.0f);
          h1[((size_t)e * NTOK + s) * HID + h] = (__bf16)v;
        }
      }
    }
  }
}

// ---------------------------------------------------------------------------
// GEMM2: out[tok][h] += gate * relu( h1[e][slot] @ W2t[e]^T + b2[e] )
__global__ __launch_bounds__(256) void moe_gemm2(
    const __bf16* __restrict__ h1, const __bf16* __restrict__ w2t,
    const float* __restrict__ b2, const int* __restrict__ counts,
    const int* __restrict__ tokens, const float* __restrict__ gates,
    float* __restrict__ out) {
  int e  = blockIdx.y;
  int cnt = counts[e];
  int m0 = blockIdx.z * 128;
  if (m0 >= cnt) return;
  int n0 = blockIdx.x * 128;

  __shared__ __align__(16) __bf16 As[2][128][KP];
  __shared__ __align__(16) __bf16 Bs[2][128][KP];
  __shared__ int   tokLds[128];
  __shared__ float gateLds[128];

  int tid = threadIdx.x;
  int lane = tid & 31, wave = tid >> 5;
  int waveM = wave & 3, waveN = wave >> 2;
  int l15 = lane & 15, half = lane >> 4;

  if (tid < 128) {
    int slot = m0 + tid;
    tokLds[tid]  = (slot < cnt) ? tokens[e * NTOK + slot] : 0;
    gateLds[tid] = (slot < cnt) ? gates[e * NTOK + slot] : 0.0f;
  }
  __syncthreads();

  v8f acc[2][4];
  #pragma unroll
  for (int i = 0; i < 2; i++)
    #pragma unroll
    for (int j = 0; j < 4; j++) acc[i][j] = (v8f)(0.0f);

  // rows past cnt read stale h1 (within allocation); their outputs are discarded
  const __bf16* Abase = h1 + ((size_t)e * NTOK + m0) * HID;
  const __bf16* Bbase = w2t + (size_t)e * HID * HID + (size_t)n0 * HID;

  int row0 = tid >> 2, row1 = row0 + 64;
  int kc = (tid & 3) * 8;
  const __bf16* gA0 = Abase + (size_t)row0 * HID + kc;
  const __bf16* gA1 = Abase + (size_t)row1 * HID + kc;
  const __bf16* gB0 = Bbase + (size_t)row0 * HID + kc;
  const __bf16* gB1 = Bbase + (size_t)row1 * HID + kc;
  uint32_t lA0[2], lA1[2], lB0[2], lB1[2];
  #pragma unroll
  for (int b = 0; b < 2; b++) {
    lA0[b] = lds_addr(&As[b][row0][kc]);
    lA1[b] = lds_addr(&As[b][row1][kc]);
    lB0[b] = lds_addr(&Bs[b][row0][kc]);
    lB1[b] = lds_addr(&Bs[b][row1][kc]);
  }

  auto issue = [&](auto bufc, auto offc) {
    constexpr int B = decltype(bufc)::value;
    constexpr int O = decltype(offc)::value;
    async_ld_b128<O>(lA0[B], gA0);  async_ld_b128<O>(lA1[B], gA1);
    async_ld_b128<O>(lB0[B], gB0);  async_ld_b128<O>(lB1[B], gB1);
  };
  auto advance = [&]() { gA0 += 64; gA1 += 64; gB0 += 64; gB1 += 64; };

  auto compute = [&](auto bufc) {
    constexpr int B = decltype(bufc)::value;
    v16bf afrag[2], bfrag[4];
    #pragma unroll
    for (int mf = 0; mf < 2; mf++) {
      int row = waveM * 32 + mf * 16 + l15;
      int kb = half * 8;
      afrag[mf] = cat16(*(const v8bf*)(&As[B][row][kb]),
                        *(const v8bf*)(&As[B][row][kb + 16]));
    }
    #pragma unroll
    for (int nf = 0; nf < 4; nf++) {
      int col = waveN * 64 + nf * 16 + l15;
      int kb = half * 8;
      bfrag[nf] = cat16(*(const v8bf*)(&Bs[B][col][kb]),
                        *(const v8bf*)(&Bs[B][col][kb + 16]));
    }
    #pragma unroll
    for (int mf = 0; mf < 2; mf++)
      #pragma unroll
      for (int nf = 0; nf < 4; nf++)
        acc[mf][nf] = __builtin_amdgcn_wmma_f32_16x16x32_bf16(
            false, afrag[mf], false, bfrag[nf], (short)0, acc[mf][nf],
            false, false);
  };

  using c0 = std::integral_constant<int, 0>;
  using c1 = std::integral_constant<int, 1>;
  using o0 = std::integral_constant<int, 0>;
  using o64 = std::integral_constant<int, 64>;

  issue(c0{}, o0{});
  issue(c1{}, o64{});
  advance();

  for (int it = 0; it < (HID / 32 - 2) / 2; ++it) {
    WAIT_ASYNC(4);  __syncthreads();
    compute(c0{});
    __syncthreads();
    issue(c0{}, o0{});
    WAIT_ASYNC(4);  __syncthreads();
    compute(c1{});
    __syncthreads();
    issue(c1{}, o64{});
    advance();
  }
  WAIT_ASYNC(4);  __syncthreads();
  compute(c0{});
  __syncthreads();
  WAIT_ASYNC(0);  __syncthreads();
  compute(c1{});

  // epilogue: bias + relu, gate-weighted scatter (exactly 2 adds/element -> deterministic)
  #pragma unroll
  for (int mf = 0; mf < 2; mf++) {
    int lbase = waveM * 32 + mf * 16 + half * 8;
    #pragma unroll
    for (int nf = 0; nf < 4; nf++) {
      int h = n0 + waveN * 64 + nf * 16 + l15;
      float bv = b2[e * HID + h];
      #pragma unroll
      for (int r = 0; r < 8; r++) {
        int lrow = lbase + r;
        if (m0 + lrow < cnt) {
          float v = fmaxf(acc[mf][nf][r] + bv, 0.0f);
          int tok = tokLds[lrow];
          float g = gateLds[lrow];
          atomicAdd(&out[(size_t)tok * HID + h], v * g);
        }
      }
    }
  }
}

// ---------------------------------------------------------------------------
extern "C" void kernel_launch(void* const* d_in, const int* in_sizes, int n_in,
                              void* d_out, int out_size, void* d_ws,
                              size_t ws_size, hipStream_t stream) {
  const float* x  = (const float*)d_in[0];
  const float* W1 = (const float*)d_in[1];
  const float* b1 = (const float*)d_in[2];
  const float* W2 = (const float*)d_in[3];
  const float* b2 = (const float*)d_in[4];
  const float* Wg = (const float*)d_in[5];
  const float* bg = (const float*)d_in[6];
  float* out = (float*)d_out;

  uint8_t* ws = (uint8_t*)d_ws;
  size_t off = 0;
  __bf16* xb  = (__bf16*)(ws + off); off += (size_t)NTOK * DIN * 2;
  __bf16* w1t = (__bf16*)(ws + off); off += (size_t)NEXP * DIN * HID * 2;
  __bf16* w2t = (__bf16*)(ws + off); off += (size_t)NEXP * HID * HID * 2;
  __bf16* h1  = (__bf16*)(ws + off); off += (size_t)NEXP * NTOK * HID * 2;
  int*    counts = (int*)(ws + off); off += 256;
  int*    tokens = (int*)(ws + off); off += (size_t)NEXP * NTOK * 4;
  float*  gatesp = (float*)(ws + off); off += (size_t)NEXP * NTOK * 4;

  moe_zero<<<(NTOK * HID + 255) / 256, 256, 0, stream>>>(out, counts);
  conv_x<<<(NTOK * DIN) / (4 * 256), 256, 0, stream>>>(x, xb);
  conv_w<<<dim3(HID / 32, DIN / 32, NEXP), 256, 0, stream>>>(W1, w1t, DIN, HID);
  conv_w<<<dim3(HID / 32, HID / 32, NEXP), 256, 0, stream>>>(W2, w2t, HID, HID);
  moe_gate<<<NTOK / 8, 256, 0, stream>>>(x, Wg, bg, counts, tokens, gatesp);
  moe_gemm1<<<dim3(HID / 128, NEXP, NTOK / 128), 256, 0, stream>>>(
      xb, w1t, b1, counts, tokens, h1);
  moe_gemm2<<<dim3(HID / 128, NEXP, NTOK / 128), 256, 0, stream>>>(
      h1, w2t, b2, counts, tokens, gatesp, out);
}
